// LSTM_1460288880740
// MI455X (gfx1250) — matrix operand order repaired and checked by reading, not compiled, not run on hardware
//
#include <hip/hip_runtime.h>
#include <hip/hip_bf16.h>
#include <math.h>

// ---------------------------------------------------------------------------
// LSTM on MI455X (gfx1250, wave32, WMMA).
// Persistent-kernel recurrence with bf16 WMMA (f32 accumulate), interleaved
// gate weight layout so each WG owns complete (g,i,f,o) columns, c-state in
// LDS, h ping-pong in global with one grid barrier per timestep.
// Round 4: (a) ping-pong h selected by integer offset from a single base
// pointer so A-stream loads lower to global_load_b128 (round 3's pointer
// array forced FLAT loads, which also count against DScnt and fight the LDS
// port); (b) the 24-fragment B weight tile is preloaded into registers
// explicitly before the time loop instead of relying on LICM hoisting.
// ---------------------------------------------------------------------------

typedef unsigned short u16;
typedef __attribute__((ext_vector_type(16))) __bf16 v16bf;
typedef __attribute__((ext_vector_type(8)))  float  v8f;

#define S_LEN 1024
#define BATCH 256
#define DIN   128
#define HID   256
#define KDIM  384          // DIN + HID
#define NCOL  1024         // 4 * HID, interleaved n = 4*j + gate
#define NWG   16
#define WG_COLS 64         // interleaved cols per WG -> 16 h-columns
#define WG_J    16
#define KSTEPS (KDIM / 32) // 12: chunks 0..3 = x-part, 4..11 = h-part
#define HELEMS (BATCH * HID)

union Frag16 { v16bf v; uint4 q[2]; };

__device__ __forceinline__ u16 f2bf(float f) {
  union { float f; unsigned u; } x; x.f = f;
  unsigned r = x.u + 0x7FFFu + ((x.u >> 16) & 1u);   // round-to-nearest-even
  return (u16)(r >> 16);
}
__device__ __forceinline__ float sigmoidf_(float x) {
  return 1.0f / (1.0f + __expf(-x));
}

// ---------------------------------------------------------------------------
// Prep kernels (fully parallel, rebuilt every call -> deterministic)
// ---------------------------------------------------------------------------
__global__ void prep_x_kernel(const float* __restrict__ x,
                              u16* __restrict__ xbf, int n) {
  for (int i = blockIdx.x * blockDim.x + threadIdx.x; i < n;
       i += gridDim.x * blockDim.x)
    xbf[i] = f2bf(x[i]);
}

__global__ void prep_w_kernel(const float* __restrict__ Wg, const float* __restrict__ Wi,
                              const float* __restrict__ Wf, const float* __restrict__ Wo,
                              const float* __restrict__ bg, const float* __restrict__ bi,
                              const float* __restrict__ bf_, const float* __restrict__ bo,
                              u16* __restrict__ wc, float* __restrict__ bc) {
  const int total = NCOL * KDIM;
  for (int idx = blockIdx.x * blockDim.x + threadIdx.x; idx < total;
       idx += gridDim.x * blockDim.x) {
    const int n = idx / KDIM;
    const int k = idx - n * KDIM;
    const int gate = n & 3;
    const int j = n >> 2;
    const float* W = (gate == 0) ? Wg : (gate == 1) ? Wi : (gate == 2) ? Wf : Wo;
    wc[(size_t)n * KDIM + k] = f2bf(W[(size_t)k * HID + j]);  // W^T, interleaved
    if (k == 0) {
      const float* bb = (gate == 0) ? bg : (gate == 1) ? bi : (gate == 2) ? bf_ : bo;
      bc[n] = bb[j];
    }
  }
}

__global__ void prep_zero_kernel(unsigned* __restrict__ h01, unsigned* __restrict__ bar) {
  const int i = blockIdx.x * blockDim.x + threadIdx.x;
  if (i < (2 * BATCH * HID) / 2) h01[i] = 0;   // h ping+pong contiguous, 65536 dwords
  if (i == 0) *bar = 0;                        // barrier counter
}

// ---------------------------------------------------------------------------
// Persistent recurrence kernel: grid = 16 WGs x 256 threads (8 waves).
// ---------------------------------------------------------------------------
__global__ __launch_bounds__(256, 1)
void lstm_persistent(const u16* __restrict__ xbf,    // [S][B][D] bf16
                     const u16* __restrict__ wc,     // [NCOL][KDIM] bf16 (W^T, interleaved)
                     const float* __restrict__ bc,   // [NCOL] bias
                     u16* __restrict__ hbase,        // [2][B][H] bf16 ping-pong
                     float* __restrict__ hfin,       // [B][H] f32 final h
                     unsigned* __restrict__ bar) {
  __shared__ __align__(16) float ldsG[128 * 68];   // gate staging (half the rows, padded)
  __shared__ __align__(16) float ldsC[256 * 16];   // persistent cell state c
  __shared__ float ldsBias[64];

  const int tid  = threadIdx.x;
  const int wg   = blockIdx.x;         // 0..15
  const int w    = tid >> 5;           // wave 0..7
  const int lane = tid & 31;
  const int l16  = lane & 15;
  const int lhi  = lane >> 4;          // 0/1
  const int wRow = (w & 3) * 64;       // wave row base (batch rows)
  const int wCol = (w >> 2) * 32;      // wave col base inside WG tile
  const int wgColBase = wg * WG_COLS;

  if (tid < 64) ldsBias[tid] = bc[wgColBase + tid];
  for (int i = tid; i < 256 * 16; i += 256) ldsC[i] = 0.f;
  __syncthreads();

  const v8f vzero = {0.f, 0.f, 0.f, 0.f, 0.f, 0.f, 0.f, 0.f};

  // Explicit B-tile preload: whole per-WG weight tile lives in 192 VGPRs for
  // all 1024 steps (no reliance on LICM hoisting).
  Frag16 bT[KSTEPS][2];
#pragma unroll
  for (int kk = 0; kk < KSTEPS; ++kk)
#pragma unroll
    for (int ci = 0; ci < 2; ++ci) {
      const int col = wgColBase + wCol + ci * 16 + l16;
      // 16-bit B layout: lane holds col N, K{k..k+7, k+16..k+23} (lanes>=16: +8).
      const uint4* q = (const uint4*)(wc + (size_t)col * KDIM + kk * 32 + lhi * 8);
      bT[kk][ci].q[0] = q[0];
      bT[kk][ci].q[1] = q[2];
    }
  __builtin_amdgcn_sched_barrier(0);

  // Load one 32-K A chunk (4 row-tiles) from `base` ([row][rstride] bf16).
  // 16-bit A layout: lanes 0-15 hold K{k..k+7, k+16..k+23}, lanes 16-31 +8.
  auto loadChunk = [&](const u16* base, int rstride, int kOff, Frag16 dst[4]) {
#pragma unroll
    for (int ri = 0; ri < 4; ++ri) {
      const int row = wRow + ri * 16 + l16;
      const uint4* q = (const uint4*)(base + (size_t)row * rstride + kOff + lhi * 8);
      dst[ri].q[0] = q[0];
      dst[ri].q[1] = q[2];          // +32B = +16 K elements
    }
  };

  // Depth-4 A-fragment ring, persistent across timesteps (x-chunks of step
  // t+1 are issued before the grid barrier of step t).
  Frag16 aF[4][4];
#pragma unroll
  for (int c = 0; c < 4; ++c) loadChunk(xbf, DIN, c * 32, aF[c]);
  __builtin_amdgcn_sched_barrier(0);

  for (int t = 0; t < S_LEN; ++t) {
    // Ping-pong selected by integer offset from ONE base pointer -> the
    // backend can prove global address space (global_load_b128, not flat).
    const u16* hcur  = hbase + (size_t)(t & 1) * HELEMS;
    u16*       hnext = hbase + (size_t)((t + 1) & 1) * HELEMS;

    v8f acc[4][2];
#pragma unroll
    for (int ri = 0; ri < 4; ++ri)
#pragma unroll
      for (int ci = 0; ci < 2; ++ci) acc[ri][ci] = vzero;

    // ---- GEMM: pre[256 x 64] += [x_t | h_{t-1}] @ Wc^T tile ----
    // Chunk kk is consumed from slot kk&3; chunk kk+4 (h-part, kOff = kk*32)
    // is loaded into the freed slot right after, then a sched fence pins the
    // order so the pipeline survives instruction scheduling.
#pragma unroll
    for (int kk = 0; kk < KSTEPS; ++kk) {
      Frag16* aCur = aF[kk & 3];
#pragma unroll
      for (int ri = 0; ri < 4; ++ri)
#pragma unroll
        for (int ci = 0; ci < 2; ++ci)
          acc[ri][ci] = __builtin_amdgcn_wmma_f32_16x16x32_bf16(
              false, aCur[ri].v, false, bT[kk][ci].v, (short)0, acc[ri][ci], false, false);

      if (kk < 8)  // h-part chunk kk+4: kb' = (kk+4)*32, kOff = kb' - 128 = kk*32
        loadChunk(hcur, HID, kk * 32, aF[kk & 3]);
      __builtin_amdgcn_sched_barrier(0);
    }

    // x-part chunks 0..3 of step t+1: no h dependency, issue now so the loads
    // fly during the elementwise phase and the grid-barrier spin.
    if (t + 1 < S_LEN) {
      const u16* xnext = xbf + (size_t)(t + 1) * (BATCH * DIN);
#pragma unroll
      for (int c = 0; c < 4; ++c) loadChunk(xnext, DIN, c * 32, aF[c]);
    }
    __builtin_amdgcn_sched_barrier(0);

    // ---- Elementwise gate update, two row-phases through 34KB LDS staging ----
#pragma unroll
    for (int phase = 0; phase < 2; ++phase) {
      if ((wRow >> 7) == phase) {
#pragma unroll
        for (int ri = 0; ri < 4; ++ri)
#pragma unroll
          for (int ci = 0; ci < 2; ++ci) {
            const int col = wCol + ci * 16 + l16;
            const int rbase = (wRow & 127) + ri * 16 + 8 * lhi;
#pragma unroll
            for (int r = 0; r < 8; ++r)
              ldsG[(rbase + r) * 68 + col] = acc[ri][ci][r];
          }
      }
      __syncthreads();
      {
        const int bh = tid & 127;
        const int jh = tid >> 7;
        const int b  = phase * 128 + bh;
        union { u16 u[8]; uint4 q; } hp;
#pragma unroll
        for (int jj = 0; jj < 8; ++jj) {
          const int jl = jh * 8 + jj;
          const float4 gv = *(const float4*)&ldsG[bh * 68 + jl * 4];
          const float g  = tanhf(gv.x + ldsBias[4 * jl + 0]);
          const float ig = sigmoidf_(gv.y + ldsBias[4 * jl + 1]);
          const float fg = sigmoidf_(gv.z + ldsBias[4 * jl + 2]);
          const float og = sigmoidf_(gv.w + ldsBias[4 * jl + 3]);
          const float cv = g * ig + ldsC[b * 16 + jl] * fg;
          const float hv = tanhf(cv) * og;
          ldsC[b * 16 + jl] = cv;
          hp.u[jj] = f2bf(hv);
          if (t == S_LEN - 1) hfin[(size_t)b * HID + wg * WG_J + jl] = hv;
        }
        *(uint4*)(hnext + (size_t)b * HID + wg * WG_J + jh * 8) = hp.q;
      }
      __syncthreads();
    }

    // ---- grid-wide barrier: make hnext visible before next step's reads ----
    __threadfence();
    __syncthreads();
    if (tid == 0) {
      __hip_atomic_fetch_add(bar, 1u, __ATOMIC_RELEASE, __HIP_MEMORY_SCOPE_AGENT);
      const unsigned target = (unsigned)(t + 1) * (unsigned)NWG;
      while (__hip_atomic_load(bar, __ATOMIC_ACQUIRE, __HIP_MEMORY_SCOPE_AGENT) < target)
        __builtin_amdgcn_s_sleep(2);
    }
    __syncthreads();
    __threadfence();
  }
}

// ---------------------------------------------------------------------------
// Logits epilogue: [256,256] @ [256,10] + bias (trivial)
// ---------------------------------------------------------------------------
__global__ __launch_bounds__(256, 1)
void logits_kernel(const float* __restrict__ hfin, const float* __restrict__ Woh,
                   const float* __restrict__ bl, float* __restrict__ out) {
  const int b = threadIdx.x;
  float acc[10];
#pragma unroll
  for (int c = 0; c < 10; ++c) acc[c] = bl[c];
  for (int k = 0; k < HID; ++k) {
    const float hv = hfin[(size_t)b * HID + k];
#pragma unroll
    for (int c = 0; c < 10; ++c) acc[c] = fmaf(hv, Woh[k * 10 + c], acc[c]);
  }
#pragma unroll
  for (int c = 0; c < 10; ++c) out[b * 10 + c] = acc[c];
}

// ---------------------------------------------------------------------------
extern "C" void kernel_launch(void* const* d_in, const int* in_sizes, int n_in,
                              void* d_out, int out_size, void* d_ws, size_t ws_size,
                              hipStream_t stream) {
  const float* x   = (const float*)d_in[0];
  const float* Wg  = (const float*)d_in[1];
  const float* Wi  = (const float*)d_in[2];
  const float* Wf  = (const float*)d_in[3];
  const float* Wo  = (const float*)d_in[4];
  const float* bg  = (const float*)d_in[5];
  const float* bi  = (const float*)d_in[6];
  const float* bf_ = (const float*)d_in[7];
  const float* bo  = (const float*)d_in[8];
  const float* Woh = (const float*)d_in[9];
  const float* bol = (const float*)d_in[10];

  char* ws = (char*)d_ws;
  size_t off = 0;
  u16* xbf = (u16*)(ws + off);   off += (size_t)S_LEN * BATCH * DIN * 2;  // 64 MB
  u16* wc  = (u16*)(ws + off);   off += (size_t)NCOL * KDIM * 2;          // 768 KB
  float* bc = (float*)(ws + off); off += (size_t)NCOL * 4;
  u16* hbase = (u16*)(ws + off); off += (size_t)2 * BATCH * HID * 2;      // ping+pong
  float* hfin = (float*)(ws + off); off += (size_t)BATCH * HID * 4;
  unsigned* bar = (unsigned*)(ws + off); off += 256;

  prep_x_kernel<<<4096, 256, 0, stream>>>(x, xbf, S_LEN * BATCH * DIN);
  prep_w_kernel<<<768, 256, 0, stream>>>(Wg, Wi, Wf, Wo, bg, bi, bf_, bo, wc, bc);
  prep_zero_kernel<<<256, 256, 0, stream>>>((unsigned*)hbase, bar);

  lstm_persistent<<<NWG, 256, 0, stream>>>(xbf, wc, bc, hbase, hfin, bar);

  logits_kernel<<<1, 256, 0, stream>>>(hfin, Woh, bol, (float*)d_out);
}
